// RetinaNet_47605417509484
// MI455X (gfx1250) — matrix operand order repaired and checked by reading, not compiled, not run on hardware
//
#include <hip/hip_runtime.h>
#include <hip/hip_bf16.h>
#include <math.h>

// ---------------- config (mirrors reference) ----------------
#define BATCH      2
#define CIN        256
#define KSTEP      32             // WMMA bf16 K per step
#define MTILE      64             // pixels per workgroup tile
#define NTILE      64             // output channels per workgroup tile
#define NCLS       80
#define NANCH      9
#define CAP        20000          // candidate cap per image
#define SCORE_TH   0.05f
#define NMS_TH     0.5f
#define MAXDET     100
#define CLS_OFFSET 10000.0f
#define SCALE_CLAMP 4.135166556742356f  // log(1000/16)

typedef __attribute__((ext_vector_type(16))) __bf16 v16bf;
typedef __attribute__((ext_vector_type(8)))  __bf16 v8bf;
typedef __attribute__((ext_vector_type(8)))  float  v8f;

__device__ __forceinline__ __bf16 f2bf(float f) {
  unsigned u = __builtin_bit_cast(unsigned, f);
  unsigned r = u + 0x7FFFu + ((u >> 16) & 1u);   // round-to-nearest-even
  unsigned short h = (unsigned short)(r >> 16);
  return __builtin_bit_cast(__bf16, h);
}
__device__ __forceinline__ float bf2f(__bf16 h) {
  unsigned short u = __builtin_bit_cast(unsigned short, h);
  unsigned v = (unsigned)u << 16;
  return __builtin_bit_cast(float, v);
}
__device__ __forceinline__ __bf16 bf0() {
  return __builtin_bit_cast(__bf16, (unsigned short)0);
}

// Pin the per-tap schedule: keep 6-12 loads in flight across every WMMA pair
// (VMEM-read group mask 0x020, MFMA/WMMA group mask 0x008). Best-effort; the
// scheduler degrades gracefully if a group can't be met.
__device__ __forceinline__ void sched_tap_pattern() {
#if __has_builtin(__builtin_amdgcn_sched_group_barrier)
  __builtin_amdgcn_sched_group_barrier(0x020, 12, 0);    // loads: chunks 0..1
  #pragma unroll
  for (int i = 0; i < 6; ++i) {
    __builtin_amdgcn_sched_group_barrier(0x008, 2, 0);   // wmma: chunk i
    __builtin_amdgcn_sched_group_barrier(0x020, 6, 0);   // loads: chunk i+2
  }
  __builtin_amdgcn_sched_group_barrier(0x008, 2, 0);     // wmma: chunk 6
  __builtin_amdgcn_sched_group_barrier(0x008, 2, 0);     // wmma: chunk 7
#endif
}

struct Acc2 { v8f a0, a1; };

// ---------------------------------------------------------------------------
// LDS-free implicit-GEMM conv tile, 64(pixel) x 64(cout) per workgroup.
// NHWC bf16 activations + [tap][coutPad][cin] bf16 weights make every lane's
// WMMA fragment a contiguous aligned global run (b128 loads, immediate
// offsets). Halo handling is a branchless per-tap base-pointer select against
// a zeroed scratch page. Depth-2 software pipeline per tap with a pinned
// VMEM/WMMA interleave so WMMA waits are partial, not full loadcnt drains.
// No LDS, no barriers.
// ---------------------------------------------------------------------------
__device__ __forceinline__ Acc2 conv_tile_accum(
    const __bf16* __restrict__ in,    // [BATCH, H, W, CIN] bf16 (NHWC)
    const __bf16* __restrict__ wk,    // [9, CoutPad, CIN]  bf16
    const char*   __restrict__ zp,    // >=1KB zero page
    int H, int W, int CoutPad, int p0, int c0)
{
  const int t    = threadIdx.x;
  const int lane = t & 31;
  const int wave = t >> 5;       // 0..7
  const int wm   = wave & 3;     // M sub-tile (4 x 16 pixels)
  const int wn   = wave >> 2;    // N sub-tile (2 x 32 couts; 2 accs each)
  const int HWp  = H * W;
  const int P    = BATCH * HWp;

  // ---- per-lane A state: this lane's pixel + K-half ----
  const int aKb = (lane >> 4) * 32;            // byte offset of K base (16 elems)
  const int pm  = p0 + wm * 16 + (lane & 15);  // pixel (M row)
  const bool pv = (pm < P);
  int n = 0, y = 0, x = 0;
  if (pv) { n = pm / HWp; int rem = pm - n * HWp; y = rem / W; x = rem - y * W; }

  const char* abase[9];
  #pragma unroll
  for (int tap = 0; tap < 9; ++tap) {
    const int dy = tap / 3 - 1, dx = tap % 3 - 1;   // compile-time
    const int iy = y + dy, ix = x + dx;
    const bool ok = pv && iy >= 0 && iy < H && ix >= 0 && ix < W;
    const char* real = (const char*)(in + ((size_t)(n * H + iy) * W + ix) * CIN);
    abase[tap] = (ok ? real : zp) + aKb;
  }

  // ---- per-lane B state: two cout columns, contiguous K run ----
  const int co0 = c0 + wn * 32 + (lane & 15);
  const int co1 = co0 + 16;
  const int bKo = (lane >> 4) * 16;            // element offset of K base
  const __bf16* b0p = wk + (size_t)co0 * CIN + bKo;
  const __bf16* b1p = wk + (size_t)co1 * CIN + bKo;

  v8f acc0 = {0.f,0.f,0.f,0.f,0.f,0.f,0.f,0.f};
  v8f acc1 = {0.f,0.f,0.f,0.f,0.f,0.f,0.f,0.f};

  union AV { v16bf v; v8bf h[2]; };

  #pragma unroll
  for (int tap = 0; tap < 9; ++tap) {
    const size_t wtoff = (size_t)tap * CoutPad * CIN;
    const char* ab = abase[tap];
    const __bf16* w0 = b0p + wtoff;
    const __bf16* w1 = b1p + wtoff;

    // depth-2 pipeline prologue: chunks 0 and 1 in flight
    AV a_s[2]; v16bf b0_s[2], b1_s[2];
    a_s[0].h[0] = *(const v8bf*)(ab);
    a_s[0].h[1] = *(const v8bf*)(ab + 16);
    b0_s[0] = *(const v16bf*)(w0);
    b1_s[0] = *(const v16bf*)(w1);
    a_s[1].h[0] = *(const v8bf*)(ab + 2 * KSTEP);
    a_s[1].h[1] = *(const v8bf*)(ab + 2 * KSTEP + 16);
    b0_s[1] = *(const v16bf*)(w0 + KSTEP);
    b1_s[1] = *(const v16bf*)(w1 + KSTEP);

    #pragma unroll
    for (int c = 0; c < 8; ++c) {
      const int sl = c & 1;
      AV a_c = a_s[sl]; v16bf b0_c = b0_s[sl], b1_c = b1_s[sl];
      if (c + 2 < 8) {                          // refill slot with chunk c+2
        const int cin2 = (c + 2) * KSTEP;
        a_s[sl].h[0] = *(const v8bf*)(ab + cin2 * 2);
        a_s[sl].h[1] = *(const v8bf*)(ab + cin2 * 2 + 16);
        b0_s[sl] = *(const v16bf*)(w0 + cin2);
        b1_s[sl] = *(const v16bf*)(w1 + cin2);
      }
      acc0 = __builtin_amdgcn_wmma_f32_16x16x32_bf16(false, a_c.v, false, b0_c,
                                                     (short)0, acc0, false, false);
      acc1 = __builtin_amdgcn_wmma_f32_16x16x32_bf16(false, a_c.v, false, b1_c,
                                                     (short)0, acc1, false, false);
    }
    sched_tap_pattern();
  }
  Acc2 r; r.a0 = acc0; r.a1 = acc1;
  return r;
}

// ---------------------------------------------------------------------------
// Generic 3x3 conv (bias + optional ReLU), NHWC bf16 in/out.
// ---------------------------------------------------------------------------
__global__ void conv3x3_wmma(const __bf16* __restrict__ in,
                             const __bf16* __restrict__ wk,
                             const float* __restrict__ bias,
                             const char* __restrict__ zp,
                             __bf16* __restrict__ out,
                             int H, int W, int Cout, int CoutPad, int relu)
{
  const int p0 = blockIdx.x * MTILE;
  const int c0 = blockIdx.y * NTILE;
  Acc2 acc = conv_tile_accum(in, wk, zp, H, W, CoutPad, p0, c0);

  const int lane = threadIdx.x & 31;
  const int wave = threadIdx.x >> 5;
  const int wm = wave & 3, wn = wave >> 2;
  const int P = BATCH * H * W;
  const int co0 = c0 + wn * 32 + (lane & 15);
  const int co1 = co0 + 16;
  #pragma unroll
  for (int r = 0; r < 8; ++r) {
    int p = p0 + wm * 16 + r + (lane >> 4) * 8;   // C/D layout: VGPR r -> M=r / 8+r
    if (p < P) {
      if (co0 < Cout) {
        float v = acc.a0[r] + bias[co0];
        if (relu) v = fmaxf(v, 0.f);
        out[(size_t)p * Cout + co0] = f2bf(v);    // NHWC: cout-contiguous stores
      }
      if (co1 < Cout) {
        float v = acc.a1[r] + bias[co1];
        if (relu) v = fmaxf(v, 0.f);
        out[(size_t)p * Cout + co1] = f2bf(v);
      }
    }
  }
}

// ---------------------------------------------------------------------------
// cls_score conv with fused sigmoid + threshold + box decode + candidate
// append (720-channel logits never touch memory).
// ---------------------------------------------------------------------------
__device__ __forceinline__ void emit_candidate(
    float logit, int p, int co, int HWp, int W,
    const __bf16* __restrict__ deltas,   // NHWC [BATCH,H,W,36]
    const float* __restrict__ anchors,
    float* cand_score, float* cand_box, float* cand_cls, int* cand_cnt)
{
  float s = 1.f / (1.f + expf(-logit));
  if (s > SCORE_TH) {
    int n = p / HWp, rem = p - n * HWp;
    int a = co / NCLS, cls = co - a * NCLS;       // channel = a*80 + cls
    const float* an = anchors + 4 * (rem * NANCH + a);
    const __bf16* dp = deltas + (size_t)p * 36 + a * 4;   // 4 contiguous bf16
    float d0 = bf2f(dp[0]), d1 = bf2f(dp[1]);
    float d2 = bf2f(dp[2]), d3 = bf2f(dp[3]);
    float aw = an[2] - an[0], ah = an[3] - an[1];
    float cx = an[0] + 0.5f * aw, cy = an[1] + 0.5f * ah;
    float dw = fminf(d2, SCALE_CLAMP), dh = fminf(d3, SCALE_CLAMP);
    float px = d0 * aw + cx, py = d1 * ah + cy;
    float pw = expf(dw) * aw, ph = expf(dh) * ah;
    int slot = atomicAdd(&cand_cnt[n], 1);
    if (slot < CAP) {
      int base = n * CAP + slot;
      cand_score[base] = s;
      cand_cls[base]   = (float)cls;
      float* bp = cand_box + (size_t)base * 4;
      bp[0] = px - 0.5f * pw;  bp[1] = py - 0.5f * ph;
      bp[2] = px + 0.5f * pw;  bp[3] = py + 0.5f * ph;
    }
  }
}

__global__ void cls_head_wmma(const __bf16* __restrict__ in,
                              const __bf16* __restrict__ wk,
                              const float* __restrict__ bias,
                              const char* __restrict__ zp,
                              const __bf16* __restrict__ deltas,   // NHWC bf16
                              const float* __restrict__ anchors,   // [H*W*9,4]
                              float* __restrict__ cand_score,
                              float* __restrict__ cand_box,
                              float* __restrict__ cand_cls,
                              int*   __restrict__ cand_cnt,
                              int H, int W, int CoutPad)
{
  const int Cout = NANCH * NCLS;  // 720
  const int p0 = blockIdx.x * MTILE;
  const int c0 = blockIdx.y * NTILE;
  Acc2 acc = conv_tile_accum(in, wk, zp, H, W, CoutPad, p0, c0);

  const int lane = threadIdx.x & 31;
  const int wave = threadIdx.x >> 5;
  const int wm = wave & 3, wn = wave >> 2;
  const int HWp = H * W, P = BATCH * HWp;
  const int co0 = c0 + wn * 32 + (lane & 15);
  const int co1 = co0 + 16;
  #pragma unroll
  for (int r = 0; r < 8; ++r) {
    int p = p0 + wm * 16 + r + (lane >> 4) * 8;
    if (p < P) {
      if (co0 < Cout)
        emit_candidate(acc.a0[r] + bias[co0], p, co0, HWp, W, deltas, anchors,
                       cand_score, cand_box, cand_cls, cand_cnt);
      if (co1 < Cout)
        emit_candidate(acc.a1[r] + bias[co1], p, co1, HWp, W, deltas, anchors,
                       cand_score, cand_box, cand_cls, cand_cnt);
    }
  }
}

// ---------------------------------------------------------------------------
// Greedy class-offset batched NMS: one block per image.
// ---------------------------------------------------------------------------
__global__ void nms_kernel(float* __restrict__ cand_score,
                           const float* __restrict__ cand_box,
                           const float* __restrict__ cand_cls,
                           const int* __restrict__ cand_cnt,
                           float* __restrict__ out)
{
  __shared__ float sv[256];
  __shared__ int   si[256];
  __shared__ float selb[4];
  __shared__ float selc, sels;
  __shared__ int   seli;

  const int n = blockIdx.x;
  const int t = threadIdx.x;
  int m = cand_cnt[n]; if (m > CAP) m = CAP;

  float*       sc = cand_score + n * CAP;
  const float* bx = cand_box  + (size_t)n * CAP * 4;
  const float* cl = cand_cls  + n * CAP;
  float* ob = out + (size_t)n * MAXDET * 4;                  // boxes  [2,100,4]
  float* os = out + (size_t)BATCH * MAXDET * 4 + n * MAXDET; // scores [2,100]
  float* oc = out + (size_t)BATCH * MAXDET * 5 + n * MAXDET; // cls    [2,100]

  for (int det = 0; det < MAXDET; ++det) {
    float best = -1.f; int bi = -1;
    for (int i = t; i < m; i += 256) {
      float v = sc[i];
      if (v > best) { best = v; bi = i; }
    }
    sv[t] = best; si[t] = bi;
    __syncthreads();
    for (int s = 128; s > 0; s >>= 1) {
      if (t < s && sv[t + s] > sv[t]) { sv[t] = sv[t + s]; si[t] = si[t + s]; }
      __syncthreads();
    }
    if (t == 0) {
      int bidx = si[0]; float bs = sv[0];
      if (bidx < 0 || bs <= 0.f) {
        seli = -1; sels = 0.f; selc = 0.f;
        selb[0] = selb[1] = selb[2] = selb[3] = 0.f;
      } else {
        seli = bidx; sels = bs; selc = cl[bidx];
        selb[0] = bx[bidx * 4 + 0]; selb[1] = bx[bidx * 4 + 1];
        selb[2] = bx[bidx * 4 + 2]; selb[3] = bx[bidx * 4 + 3];
      }
      ob[det * 4 + 0] = selb[0]; ob[det * 4 + 1] = selb[1];
      ob[det * 4 + 2] = selb[2]; ob[det * 4 + 3] = selb[3];
      os[det] = sels; oc[det] = selc;
    }
    __syncthreads();
    if (seli >= 0) {
      float off = selc * CLS_OFFSET;
      float sx0 = selb[0] + off, sy0 = selb[1] + off;
      float sx1 = selb[2] + off, sy1 = selb[3] + off;
      float sarea = (sx1 - sx0) * (sy1 - sy0);
      for (int i = t; i < m; i += 256) {
        if (sc[i] <= 0.f) continue;
        float o2 = cl[i] * CLS_OFFSET;
        float x0 = bx[i * 4 + 0] + o2, y0 = bx[i * 4 + 1] + o2;
        float x1 = bx[i * 4 + 2] + o2, y1 = bx[i * 4 + 3] + o2;
        float lx = fmaxf(sx0, x0), ly = fmaxf(sy0, y0);
        float rx = fminf(sx1, x1), ry = fminf(sy1, y1);
        float iw = fmaxf(rx - lx, 0.f), ih = fmaxf(ry - ly, 0.f);
        float inter = iw * ih;
        float area  = (x1 - x0) * (y1 - y0);
        float iou = inter / (sarea + area - inter + 1e-9f);
        if (iou >= NMS_TH) sc[i] = 0.f;   // suppresses self too
      }
    }
    __syncthreads();
  }
}

// ---------------------------------------------------------------------------
// One-time helpers.
// ---------------------------------------------------------------------------
// NCHW fp32 -> NHWC bf16
__global__ void cvt_nhwc(const float* __restrict__ in, __bf16* __restrict__ out,
                         int HWp, int nElem) {
  int i = blockIdx.x * 256 + threadIdx.x;        // over P*CIN (NHWC order)
  if (i >= nElem) return;
  int c = i & (CIN - 1), pix = i >> 8;
  int n = pix / HWp, rem = pix - n * HWp;
  out[i] = f2bf(in[((size_t)(n * CIN + c)) * HWp + rem]);
}

// OIHW fp32 -> [tap][coutPad][cin] bf16 (zero-padded couts)
__global__ void repack_w(const float* __restrict__ w, __bf16* __restrict__ wk,
                         int Cout, int CoutPad) {
  int i = blockIdx.x * 256 + threadIdx.x;        // over 9*CoutPad*CIN (dst order)
  int tot = 9 * CoutPad * CIN;
  if (i >= tot) return;
  int cin  = i & (CIN - 1);
  int rest = i >> 8;                             // tap*CoutPad + co
  int tap  = rest / CoutPad, co = rest - tap * CoutPad;
  wk[i] = (co < Cout) ? f2bf(w[((size_t)co * CIN + cin) * 9 + tap]) : bf0();
}

__global__ void reset_kernel(int* cnt, float* zp) {
  int t = threadIdx.x;
  if (t < BATCH) cnt[t] = 0;
  zp[t] = 0.f;                                    // 256 threads -> 1KB zero page
}

// ---------------------------------------------------------------------------
// Host orchestration. Input order = setup_inputs() flattened (JAX sorted-key
// dict order for params): feat0..4, bbox_b*4, bbox_pred_b, bbox_pred_w,
// bbox_w*4, cls_b*4, cls_score_b, cls_score_w, cls_w*4, anchors lvl0..4.
// ---------------------------------------------------------------------------
extern "C" void kernel_launch(void* const* d_in, const int* in_sizes, int n_in,
                              void* d_out, int out_size, void* d_ws, size_t ws_size,
                              hipStream_t stream) {
  (void)in_sizes; (void)n_in; (void)out_size; (void)ws_size;

  const float* feats[5];
  for (int l = 0; l < 5; ++l) feats[l] = (const float*)d_in[l];
  const float* bbox_b[4];   for (int j = 0; j < 4; ++j) bbox_b[j] = (const float*)d_in[5 + j];
  const float* bbox_pred_b = (const float*)d_in[9];
  const float* bbox_pred_w = (const float*)d_in[10];
  const float* bbox_w[4];   for (int j = 0; j < 4; ++j) bbox_w[j] = (const float*)d_in[11 + j];
  const float* cls_b[4];    for (int j = 0; j < 4; ++j) cls_b[j] = (const float*)d_in[15 + j];
  const float* cls_score_b = (const float*)d_in[19];
  const float* cls_score_w = (const float*)d_in[20];
  const float* cls_w[4];    for (int j = 0; j < 4; ++j) cls_w[j] = (const float*)d_in[21 + j];
  const float* anchors[5];  for (int l = 0; l < 5; ++l) anchors[l] = (const float*)d_in[25 + l];

  static const int HH[5] = {100, 50, 25, 13, 7};
  static const int WW[5] = {100, 50, 25, 13, 7};

  // ---- carve workspace ----
  char* wp = (char*)d_ws;
  auto carve = [&](size_t bytes) -> char* {
    char* r = wp;
    wp += (bytes + 255) & ~(size_t)255;
    return r;
  };
  __bf16* featb[5];
  for (int l = 0; l < 5; ++l)
    featb[l] = (__bf16*)carve((size_t)BATCH * HH[l] * WW[l] * CIN * sizeof(__bf16));
  __bf16* buf0 = (__bf16*)carve((size_t)BATCH * 100 * 100 * CIN * sizeof(__bf16));
  __bf16* buf1 = (__bf16*)carve((size_t)BATCH * 100 * 100 * CIN * sizeof(__bf16));
  __bf16* deltas[5];
  for (int l = 0; l < 5; ++l)
    deltas[l] = (__bf16*)carve((size_t)BATCH * HH[l] * WW[l] * 36 * sizeof(__bf16));
  __bf16* wk_bbox[4], *wk_cls[4];
  for (int j = 0; j < 4; ++j) {
    wk_bbox[j] = (__bf16*)carve((size_t)9 * 256 * CIN * sizeof(__bf16));
    wk_cls[j]  = (__bf16*)carve((size_t)9 * 256 * CIN * sizeof(__bf16));
  }
  __bf16* wk_pred  = (__bf16*)carve((size_t)9 * 64  * CIN * sizeof(__bf16));   // 36 -> pad 64
  __bf16* wk_score = (__bf16*)carve((size_t)9 * 768 * CIN * sizeof(__bf16));   // 720 -> pad 768
  float* zpage      = (float*)carve(1024);
  float* cand_score = (float*)carve((size_t)BATCH * CAP * sizeof(float));
  float* cand_box   = (float*)carve((size_t)BATCH * CAP * 4 * sizeof(float));
  float* cand_cls   = (float*)carve((size_t)BATCH * CAP * sizeof(float));
  int*   cand_cnt   = (int*)carve(256);

  dim3 blk(256);
  reset_kernel<<<1, 256, 0, stream>>>(cand_cnt, zpage);

  // ---- one-time converts / repacks ----
  for (int l = 0; l < 5; ++l) {
    int nEl = BATCH * HH[l] * WW[l] * CIN;
    cvt_nhwc<<<(nEl + 255) / 256, blk, 0, stream>>>(feats[l], featb[l],
                                                    HH[l] * WW[l], nEl);
  }
  auto repack = [&](const float* w, __bf16* dst, int Cout, int CoutPad) {
    int tot = 9 * CoutPad * CIN;
    repack_w<<<(tot + 255) / 256, blk, 0, stream>>>(w, dst, Cout, CoutPad);
  };
  for (int j = 0; j < 4; ++j) {
    repack(bbox_w[j], wk_bbox[j], 256, 256);
    repack(cls_w[j],  wk_cls[j],  256, 256);
  }
  repack(bbox_pred_w, wk_pred,  36,  64);
  repack(cls_score_w, wk_score, 720, 768);

  // ---- per-level head ----
  for (int l = 0; l < 5; ++l) {
    const int H = HH[l], W = WW[l];
    const int P = BATCH * H * W;
    const int gx = (P + MTILE - 1) / MTILE;

    // bbox subnet -> bbox_pred (deltas needed by cls epilogue decode)
    const __bf16* x = featb[l];
    for (int j = 0; j < 4; ++j) {
      __bf16* o = (j & 1) ? buf1 : buf0;
      dim3 g(gx, 256 / NTILE);
      conv3x3_wmma<<<g, blk, 0, stream>>>(x, wk_bbox[j], bbox_b[j], (const char*)zpage,
                                          o, H, W, 256, 256, 1);
      x = o;
    }
    {
      dim3 g(gx, 1);
      conv3x3_wmma<<<g, blk, 0, stream>>>(x, wk_pred, bbox_pred_b, (const char*)zpage,
                                          deltas[l], H, W, 36, 64, 0);
    }

    // cls subnet -> fused cls_score head
    x = featb[l];
    for (int j = 0; j < 4; ++j) {
      __bf16* o = (j & 1) ? buf1 : buf0;
      dim3 g(gx, 256 / NTILE);
      conv3x3_wmma<<<g, blk, 0, stream>>>(x, wk_cls[j], cls_b[j], (const char*)zpage,
                                          o, H, W, 256, 256, 1);
      x = o;
    }
    {
      dim3 g(gx, 768 / NTILE);
      cls_head_wmma<<<g, blk, 0, stream>>>(x, wk_score, cls_score_b, (const char*)zpage,
                                           deltas[l], anchors[l],
                                           cand_score, cand_box, cand_cls, cand_cnt,
                                           H, W, 768);
    }
  }

  nms_kernel<<<BATCH, 256, 0, stream>>>(cand_score, cand_box, cand_cls, cand_cnt,
                                        (float*)d_out);
}
